// BirdCLEFEncoder_85349590106330
// MI455X (gfx1250) — matrix operand order repaired and checked by reading, not compiled, not run on hardware
//
#include <hip/hip_runtime.h>
#include <hip/hip_bf16.h>

typedef unsigned short u16t;
typedef unsigned int   u32t;
typedef __attribute__((ext_vector_type(16))) __bf16 v16bf;
typedef __attribute__((ext_vector_type(8)))  float  v8f;
typedef __attribute__((ext_vector_type(4)))  unsigned int u32x4;
typedef __attribute__((ext_vector_type(8)))  int i32x8;
typedef __attribute__((ext_vector_type(4)))  int i32x4;

#if defined(__AMDGCN__) && defined(__has_builtin)
# if __has_builtin(__builtin_amdgcn_tensor_load_to_lds) && __has_builtin(__builtin_amdgcn_s_wait_tensorcnt)
#  define HAVE_TDM 1
# endif
#endif
#ifndef HAVE_TDM
# define HAVE_TDM 0
#endif

union FragU { uint4 q[2]; v16bf v; u16t h[16]; };

__device__ __forceinline__ u16t f2bf(float f){
  u32t u = __float_as_uint(f);
  u32t r = u + 0x7FFFu + ((u >> 16) & 1u);   // round-to-nearest-even
  return (u16t)(r >> 16);
}
__device__ __forceinline__ float gelu_f(float x){
  return 0.5f * x * (1.0f + erff(x * 0.7071067811865476f));
}
__device__ __forceinline__ v8f wmma_bf16(const v16bf& a, const v16bf& b, const v8f& c){
  return __builtin_amdgcn_wmma_f32_16x16x32_bf16(false, a, false, b, (short)0, c, false, false);
}

// ---------------- conv1: direct fp32 (Cin=1, 3x3, pad1, stride(2,1)) ----------------
__global__ __launch_bounds__(256)
void conv1_k(const float* __restrict__ in, const float* __restrict__ w,
             const float* __restrict__ bias, float* __restrict__ out){
  int idx = blockIdx.x * 256 + threadIdx.x;      // 8*32*64*1024
  int wi = idx & 1023;
  int ho = (idx >> 10) & 63;
  int co = (idx >> 16) & 31;
  int b  = idx >> 21;
  float acc = bias[co];
#pragma unroll
  for (int kh = 0; kh < 3; ++kh){
    int h = ho * 2 + kh - 1;
    if (h < 0 || h >= 128) continue;
#pragma unroll
    for (int kw = 0; kw < 3; ++kw){
      int x = wi + kw - 1;
      if (x < 0 || x >= 1024) continue;
      acc += in[(size_t)(b * 128 + h) * 1024 + x] * w[co * 9 + kh * 3 + kw];
    }
  }
  out[idx] = acc;
}

// ---------------- batchnorm stats (training mode: batch mean/var) ----------------
__global__ __launch_bounds__(256)
void bn_stats_k(const float* __restrict__ raw, float* __restrict__ mv, int C, int HW){
  int c = blockIdx.x;
  int tid = threadIdx.x;
  float s = 0.f, s2 = 0.f;
  for (int b = 0; b < 8; ++b){
    const float* base = raw + (size_t)(b * C + c) * HW;
    for (int i = tid; i < HW; i += 256){
      float v = base[i];
      s += v; s2 += v * v;
    }
  }
  __shared__ float sh[256], sh2[256];
  sh[tid] = s; sh2[tid] = s2;
  __syncthreads();
  for (int st = 128; st > 0; st >>= 1){
    if (tid < st){ sh[tid] += sh[tid + st]; sh2[tid] += sh2[tid + st]; }
    __syncthreads();
  }
  if (tid == 0){
    float n = (float)(8 * HW);
    float m = sh[0] / n;
    mv[c] = m;
    mv[C + c] = sh2[0] / n - m * m;
  }
}

// ---------------- normalize + GELU -> bf16 (optionally into token layout) ----------------
__global__ __launch_bounds__(256)
void bn_gelu_k(const float* __restrict__ raw, const float* __restrict__ mv,
               u16t* __restrict__ dst, int Cmask, int lgHW, int token_mode){
  int idx = blockIdx.x * 256 + threadIdx.x;
  int c = (idx >> lgHW) & Cmask;
  float m = mv[c], v = mv[Cmask + 1 + c];
  float y = (raw[idx] - m) * rsqrtf(v + 1e-5f);
  y = gelu_f(y);
  if (!token_mode){
    dst[idx] = f2bf(y);
  } else {
    // raw layout (B,128,8,1024) -> tokens (B,1024,1024), feature = c*8 + f
    int wi = idx & 1023;
    int f  = (idx >> 10) & 7;
    int b  = idx >> 20;
    dst[(size_t)(b * 1024 + wi) * 1024 + c * 8 + f] = f2bf(y);
  }
}

// ---------------- implicit-GEMM conv via WMMA bf16 (FS/PAD compile-time) ----------------
template<int FS, int PAD>
__global__ __launch_bounds__(256)
void conv_wmma_k(const u16t* __restrict__ Wt, const u16t* __restrict__ act,
                 const float* __restrict__ bias, float* __restrict__ out,
                 int Cout, int Cin, int Ktot, int Hin, int Hout, int ntiles){
  int tid  = blockIdx.x * 8 + (threadIdx.x >> 5);
  int lane = threadIdx.x & 31;
  int mt = tid / ntiles;
  int nt = tid - mt * ntiles;
  if (mt >= (Cout >> 4)) return;                // wave-uniform
  int colrow = lane & 15;
  int hi8    = (lane >> 4) << 3;
  int kbA    = (lane < 16) ? 0 : 16;
  int kbB    = (lane < 16) ? 0 : 16;
  int p   = nt * 16 + colrow;
  int wi  = p & 1023;
  int rem = p >> 10;
  int ho  = rem % Hout;
  int b   = rem / Hout;
  const u16t* arow = Wt + (size_t)(mt * 16 + colrow) * Ktot;
  constexpr int FS2 = FS * FS;
  v8f acc = {};
  for (int k0 = 0; k0 < Ktot; k0 += 32){
    FragU a, bf;
    a.q[0] = *(const uint4*)(arow + k0 + kbA);
    a.q[1] = *(const uint4*)(arow + k0 + kbA + 8);
#pragma unroll
    for (int i = 0; i < 16; ++i){
      int k  = k0 + kbB + i;
      int ci = k / FS2;                 // constant divisor -> mul/shift
      int r  = k - ci * FS2;
      int kh = r / FS;
      int kw = r - kh * FS;
      int hin = ho * 2 + kh - PAD;
      int win = wi + kw - PAD;
      u16t vv = 0;
      if (hin >= 0 && hin < Hin && win >= 0 && win < 1024)
        vv = act[(size_t)((b * Cin + ci) * Hin + hin) * 1024 + win];
      bf.h[i] = vv;
    }
    acc = wmma_bf16(a.v, bf.v, acc);
  }
#pragma unroll
  for (int r = 0; r < 8; ++r){
    int co = mt * 16 + r + hi8;
    out[(size_t)((b * Cout + co) * Hout + ho) * 1024 + wi] = acc[r] + bias[co];
  }
}

// ---------------- generic WMMA GEMM, 2x2 register blocking (32x32 per wave) ----------------
// mode 0: outf = val; 1: outf += val; 2: outb = bf16(gelu(val));
// mode 3: split-qkv bf16 -> q/k/v (B,NH,T,HD); mode 4: outf = val + sinusoidal PE
__global__ __launch_bounds__(256)
void gemm_wmma_k(const u16t* __restrict__ A, const u16t* __restrict__ Bt,
                 const float* __restrict__ bias,
                 float* __restrict__ outf, u16t* __restrict__ outb,
                 u16t* __restrict__ qp, u16t* __restrict__ kp, u16t* __restrict__ vp,
                 int M, int N, int K, int mode){
  int tid = blockIdx.x * 8 + (threadIdx.x >> 5);
  int nb  = N >> 5;
  int mt  = tid / nb;
  int nt  = tid - mt * nb;
  if (mt >= (M >> 5)) return;
  int lane   = threadIdx.x & 31;
  int colrow = lane & 15;
  int hi8    = (lane >> 4) << 3;
  int kbA    = (lane < 16) ? 0 : 16;
  int kbB    = (lane < 16) ? 0 : 16;
  const u16t* a0p = A  + (size_t)(mt * 32 + colrow) * K;
  const u16t* a1p = a0p + (size_t)16 * K;
  const u16t* b0p = Bt + (size_t)(nt * 32 + colrow) * K;
  const u16t* b1p = b0p + (size_t)16 * K;
  v8f c00 = {}, c01 = {}, c10 = {}, c11 = {};
  for (int k0 = 0; k0 < K; k0 += 32){
    __builtin_prefetch(a0p + k0 + 64, 0, 1);   // global_prefetch_b8, two iters ahead
    __builtin_prefetch(b0p + k0 + 64, 0, 1);
    FragU a0, a1, b0, b1;
    a0.q[0] = *(const uint4*)(a0p + k0 + kbA); a0.q[1] = *(const uint4*)(a0p + k0 + kbA + 8);
    a1.q[0] = *(const uint4*)(a1p + k0 + kbA); a1.q[1] = *(const uint4*)(a1p + k0 + kbA + 8);
    b0.q[0] = *(const uint4*)(b0p + k0 + kbB); b0.q[1] = *(const uint4*)(b0p + k0 + kbB + 8);
    b1.q[0] = *(const uint4*)(b1p + k0 + kbB); b1.q[1] = *(const uint4*)(b1p + k0 + kbB + 8);
    c00 = wmma_bf16(a0.v, b0.v, c00);
    c01 = wmma_bf16(a0.v, b1.v, c01);
    c10 = wmma_bf16(a1.v, b0.v, c10);
    c11 = wmma_bf16(a1.v, b1.v, c11);
  }
  v8f accs[4] = {c00, c01, c10, c11};
#pragma unroll
  for (int t4 = 0; t4 < 4; ++t4){
    int mi = t4 >> 1, ni = t4 & 1;
    v8f acc = accs[t4];
#pragma unroll
    for (int r = 0; r < 8; ++r){
      int row = mt * 32 + mi * 16 + r + hi8;
      int col = nt * 32 + ni * 16 + colrow;
      float val = acc[r] + bias[col];
      if (mode == 0){
        outf[(size_t)row * N + col] = val;
      } else if (mode == 1){
        outf[(size_t)row * N + col] += val;
      } else if (mode == 2){
        outb[(size_t)row * N + col] = f2bf(gelu_f(val));
      } else if (mode == 3){
        int which = col >> 8;
        int hh = (col >> 5) & 7;
        int d  = col & 31;
        int b_ = row >> 10;
        int t  = row & 1023;
        u16t* dst = (which == 0) ? qp : (which == 1) ? kp : vp;
        dst[(size_t)((b_ * 8 + hh) * 1024 + t) * 32 + d] = f2bf(val);
      } else {
        int t = row & 1023;
        float freq = __expf(-(float)(col & ~1) * (9.210340371976184f / 256.0f));
        float ang  = (float)t * freq;
        val += (col & 1) ? __cosf(ang) : __sinf(ang);
        outf[(size_t)row * N + col] = val;
      }
    }
  }
}

// ---------------- LayerNorm (H=256) -> bf16 ----------------
__global__ __launch_bounds__(256)
void layernorm_k(const float* __restrict__ x, const float* __restrict__ g,
                 const float* __restrict__ bb, u16t* __restrict__ xn){
  int row = blockIdx.x;
  int c   = threadIdx.x;
  float v = x[(size_t)row * 256 + c];
  __shared__ float sh[256];
  sh[c] = v; __syncthreads();
  for (int st = 128; st > 0; st >>= 1){ if (c < st) sh[c] += sh[c + st]; __syncthreads(); }
  float m = sh[0] * (1.0f / 256.0f);
  __syncthreads();
  float d = v - m;
  sh[c] = d * d; __syncthreads();
  for (int st = 128; st > 0; st >>= 1){ if (c < st) sh[c] += sh[c + st]; __syncthreads(); }
  float var = sh[0] * (1.0f / 256.0f);
  xn[(size_t)row * 256 + c] = f2bf(d * rsqrtf(var + 1e-5f) * g[c] + bb[c]);
}

// ---------------- attention scores: one WMMA per 16x16 tile (K = HD = 32) ----------------
__global__ __launch_bounds__(256)
void attn_scores_k(const u16t* __restrict__ q, const u16t* __restrict__ kptr,
                   float* __restrict__ sc, int mtype){
  int tid  = blockIdx.x * 8 + (threadIdx.x >> 5);   // NH*64*64 tiles
  int lane = threadIdx.x & 31;
  int st = tid & 63;
  int tt = (tid >> 6) & 63;
  int h  = tid >> 12;
  int t0 = tt << 4, s0 = st << 4;
  int colrow = lane & 15;
  int hi8    = (lane >> 4) << 3;
  float* srow = sc + (size_t)h * 1024 * 1024;
  if (mtype <= 1){
    int W = mtype ? 16 : 8;
    int dd = t0 - s0; if (dd < 0) dd = -dd;
    if (dd > W + 15){                               // tile fully masked (wave-uniform)
#pragma unroll
      for (int r = 0; r < 8; ++r)
        srow[(size_t)(t0 + r + hi8) * 1024 + s0 + colrow] = -1e9f;
      return;
    }
  }
  int kbA = (lane < 16) ? 0 : 16;
  int kbB = (lane < 16) ? 0 : 16;
  FragU a, b;
  const u16t* qa = q    + (size_t)(h * 1024 + t0 + colrow) * 32;
  const u16t* ka = kptr + (size_t)(h * 1024 + s0 + colrow) * 32;
  a.q[0] = *(const uint4*)(qa + kbA);
  a.q[1] = *(const uint4*)(qa + kbA + 8);
  b.q[0] = *(const uint4*)(ka + kbB);
  b.q[1] = *(const uint4*)(ka + kbB + 8);
  v8f acc = {};
  acc = wmma_bf16(a.v, b.v, acc);
#pragma unroll
  for (int r = 0; r < 8; ++r){
    int t = t0 + r + hi8;
    int s = s0 + colrow;
    float v = acc[r] * 0.17677669529663689f;        // 1/sqrt(32)
    bool allow;
    if (mtype == 0)      allow = (t - s <= 8)  && (s - t <= 8);
    else if (mtype == 1) allow = (t - s <= 16) && (s - t <= 16);
    else                 allow = ((t & 15) == 0) || (t == s);
    srow[(size_t)t * 1024 + s] = allow ? v : -1e9f;
  }
}

// ---------------- row softmax (1024 cols) -> bf16 probs ----------------
__global__ __launch_bounds__(256)
void softmax_k(const float* __restrict__ sc, u16t* __restrict__ pr){
  int row = blockIdx.x;
  int tid = threadIdx.x;
  const float* s = sc + (size_t)row * 1024;
  u16t* p = pr + (size_t)row * 1024;
  float v0 = s[tid], v1 = s[tid + 256], v2 = s[tid + 512], v3 = s[tid + 768];
  float mx = fmaxf(fmaxf(v0, v1), fmaxf(v2, v3));
  __shared__ float sh[256];
  sh[tid] = mx; __syncthreads();
  for (int st = 128; st > 0; st >>= 1){ if (tid < st) sh[tid] = fmaxf(sh[tid], sh[tid + st]); __syncthreads(); }
  float m = sh[0]; __syncthreads();
  v0 = __expf(v0 - m); v1 = __expf(v1 - m); v2 = __expf(v2 - m); v3 = __expf(v3 - m);
  sh[tid] = v0 + v1 + v2 + v3; __syncthreads();
  for (int st = 128; st > 0; st >>= 1){ if (tid < st) sh[tid] += sh[tid + st]; __syncthreads(); }
  float inv = 1.0f / sh[0];
  p[tid] = f2bf(v0 * inv); p[tid + 256] = f2bf(v1 * inv);
  p[tid + 512] = f2bf(v2 * inv); p[tid + 768] = f2bf(v3 * inv);
}

// ---------------- attn * V with V head staged in LDS (TDM async copy when available) ---------
// One block per head: stage V[h] (1024x32 bf16 = 64KB) into LDS, then 8 waves x 8 q-tiles.
__global__ __launch_bounds__(256)
void attn_av_k(const u16t* __restrict__ probs, const u16t* __restrict__ vptr,
               u16t* __restrict__ o, int brow0){
  __shared__ u16t vlds[1024 * 32];
  int h    = blockIdx.x;                 // 8 heads
  int wave = threadIdx.x >> 5;
  int lane = threadIdx.x & 31;
  const u16t* vhead = vptr + (size_t)h * 1024 * 32;

#if HAVE_TDM
  if (wave == 0){
    // D# per CDNA5 ISA ch.8: 2D tensor, data_size=2B, tensor 32x1024 contiguous,
    // tile 32x1024 -> plain 64KB global->LDS DMA via the Tensor Data Mover.
    unsigned long long ga = (unsigned long long)(uintptr_t)vhead;
    unsigned lds_off = (unsigned)(uintptr_t)(&vlds[0]);
    u32x4 g0 = { 1u,                                   // count=1, user mode
                 lds_off,                              // lds_addr
                 (u32t)ga,                             // global_addr[31:0]
                 (u32t)((ga >> 32) & 0x1FFFFFFu) | (2u << 30) }; // addr[56:32] | type=2
    i32x8 g1 = { (int)(1u << 16),                      // data_size=1 (2 bytes)
                 (int)(32u << 16),                     // tensor_dim0=32 (low16)
                 (int)(1024u << 16),                   // tensor_dim1=1024 (low16)
                 (int)(32u << 16),                     // tile_dim0=32
                 (int)1024,                            // tile_dim1=1024, tile_dim2=0
                 (int)32,                              // tensor_dim0_stride=32
                 0, 0 };
    i32x4 gz = { 0, 0, 0, 0 };
# if __clang_major__ >= 23
    i32x8 gz8 = { 0, 0, 0, 0, 0, 0, 0, 0 };
    __builtin_amdgcn_tensor_load_to_lds(g0, g1, gz, gz, gz8, 0);
# else
    __builtin_amdgcn_tensor_load_to_lds(g0, g1, gz, gz, 0);
# endif
    __builtin_amdgcn_s_wait_tensorcnt(0);
  }
  __syncthreads();
#else
  for (int i = threadIdx.x; i < (1024 * 32) / 8; i += 256)
    ((uint4*)vlds)[i] = ((const uint4*)vhead)[i];
  __syncthreads();
#endif

  int colrow = lane & 15;
  int hi8    = (lane >> 4) << 3;
  int kbA    = (lane < 16) ? 0 : 16;
  int kbB    = (lane < 16) ? 0 : 16;
  for (int it = 0; it < 8; ++it){
    int tt = wave * 8 + it;              // 64 q-tiles per head
    const u16t* arow = probs + (size_t)(h * 1024 + tt * 16 + colrow) * 1024;
    v8f c0 = {}, c1 = {};
    for (int k0 = 0; k0 < 1024; k0 += 32){
      FragU a, b0, b1;
      a.q[0] = *(const uint4*)(arow + k0 + kbA);
      a.q[1] = *(const uint4*)(arow + k0 + kbA + 8);
#pragma unroll
      for (int i = 0; i < 16; ++i){
        int s = k0 + kbB + i;
        b0.h[i] = vlds[s * 32 + colrow];
        b1.h[i] = vlds[s * 32 + 16 + colrow];
      }
      c0 = wmma_bf16(a.v, b0.v, c0);
      c1 = wmma_bf16(a.v, b1.v, c1);
    }
#pragma unroll
    for (int r = 0; r < 8; ++r){
      int t = tt * 16 + r + hi8;
      size_t base = (size_t)(brow0 + t) * 256 + h * 32;
      o[base + colrow]      = f2bf(c0[r]);
      o[base + 16 + colrow] = f2bf(c1[r]);
    }
  }
}

// ---------------- weight conversion helpers ----------------
__global__ __launch_bounds__(256)
void cv_bf16_k(const float* __restrict__ src, u16t* __restrict__ dst, int n){
  int idx = blockIdx.x * 256 + threadIdx.x;
  if (idx < n) dst[idx] = f2bf(src[idx]);
}
__global__ __launch_bounds__(256)
void tr_bf16_k(const float* __restrict__ src, u16t* __restrict__ dst, int K, int N){
  int idx = blockIdx.x * 256 + threadIdx.x;
  if (idx >= K * N) return;
  int n = idx % N, k = idx / N;
  dst[(size_t)n * K + k] = f2bf(src[idx]);      // (KxN) -> (NxK)
}

extern "C" void kernel_launch(void* const* d_in, const int* in_sizes, int n_in,
                              void* d_out, int out_size, void* d_ws, size_t ws_size,
                              hipStream_t stream){
  (void)in_sizes; (void)n_in; (void)ws_size;
  const float* x_in   = (const float*)d_in[0];
  const float* cw1    = (const float*)d_in[1];
  const float* cb1    = (const float*)d_in[2];
  const float* cw2    = (const float*)d_in[3];
  const float* cb2    = (const float*)d_in[4];
  const float* cw3    = (const float*)d_in[5];
  const float* cb3    = (const float*)d_in[6];
  const float* cw4    = (const float*)d_in[7];
  const float* cb4    = (const float*)d_in[8];
  const float* proj_w = (const float*)d_in[9];
  const float* proj_b = (const float*)d_in[10];
  const float* ln1_g  = (const float*)d_in[11];
  const float* ln1_b  = (const float*)d_in[12];
  const float* wqkv   = (const float*)d_in[13];
  const float* bqkv   = (const float*)d_in[14];
  const float* wo     = (const float*)d_in[15];
  const float* bo     = (const float*)d_in[16];
  const float* w1     = (const float*)d_in[17];
  const float* b1     = (const float*)d_in[18];
  const float* w2     = (const float*)d_in[19];
  const float* b2     = (const float*)d_in[20];
  const float* ln2_g  = (const float*)d_in[21];
  const float* ln2_b  = (const float*)d_in[22];

  char* ws = (char*)d_ws;
  size_t off = 0;
  auto carve = [&](size_t bytes) -> char* {
    char* p = ws + off;
    off += (bytes + 255) & ~(size_t)255;
    return p;
  };
  float* raw    = (float*)carve(16777216ull * 4);        // conv raw; later attn scores+probs
  u16t*  actA   = (u16t*) carve(16777216ull * 2);
  u16t*  actB   = (u16t*) carve(16777216ull * 2);        // also holds tokens (8192x1024 bf16)
  float* xres   = (float*)carve(8192ull * 256 * 4);      // residual stream (B,T,H)
  u16t*  xn     = (u16t*) carve(8192ull * 256 * 2);
  u16t*  qb     = (u16t*) carve(8ull * 8 * 1024 * 32 * 2);
  u16t*  kbuf   = (u16t*) carve(8ull * 8 * 1024 * 32 * 2);
  u16t*  vbuf   = (u16t*) carve(8ull * 8 * 1024 * 32 * 2);
  u16t*  obf    = (u16t*) carve(8192ull * 256 * 2);
  u16t*  hbuf   = (u16t*) carve(8192ull * 1024 * 2);
  u16t*  wc2    = (u16t*) carve(64ull  * 800  * 2);
  u16t*  wc3    = (u16t*) carve(128ull * 3136 * 2);
  u16t*  wc4    = (u16t*) carve(128ull * 6272 * 2);
  u16t*  projbt = (u16t*) carve(256ull * 1024 * 2);
  u16t*  qkvbt  = (u16t*) carve(4ull * 768 * 256 * 2);
  u16t*  wobt   = (u16t*) carve(4ull * 256 * 256 * 2);
  u16t*  w1bt   = (u16t*) carve(4ull * 1024 * 256 * 2);
  u16t*  w2bt   = (u16t*) carve(4ull * 256 * 1024 * 2);
  float* stats  = (float*)carve(256 * 4);
  float* scores = raw;                                   // 8*1024*1024 fp32 per batch
  u16t*  probs  = (u16t*)(raw + 8ull * 1024 * 1024);     // 8*1024*1024 bf16 per batch

  // ---- weight packing ----
  cv_bf16_k<<<(64 * 800 + 255) / 256, 256, 0, stream>>>(cw2, wc2, 64 * 800);
  cv_bf16_k<<<(128 * 3136 + 255) / 256, 256, 0, stream>>>(cw3, wc3, 128 * 3136);
  cv_bf16_k<<<(128 * 6272 + 255) / 256, 256, 0, stream>>>(cw4, wc4, 128 * 6272);
  tr_bf16_k<<<(1024 * 256 + 255) / 256, 256, 0, stream>>>(proj_w, projbt, 1024, 256);
  for (int k = 0; k < 4; ++k){
    tr_bf16_k<<<(256 * 768 + 255) / 256, 256, 0, stream>>>(wqkv + (size_t)k * 256 * 768, qkvbt + (size_t)k * 768 * 256, 256, 768);
    tr_bf16_k<<<(256 * 256 + 255) / 256, 256, 0, stream>>>(wo + (size_t)k * 256 * 256, wobt + (size_t)k * 256 * 256, 256, 256);
    tr_bf16_k<<<(256 * 1024 + 255) / 256, 256, 0, stream>>>(w1 + (size_t)k * 256 * 1024, w1bt + (size_t)k * 1024 * 256, 256, 1024);
    tr_bf16_k<<<(1024 * 256 + 255) / 256, 256, 0, stream>>>(w2 + (size_t)k * 1024 * 256, w2bt + (size_t)k * 256 * 1024, 1024, 256);
  }

  // ---- conv stem ----
  conv1_k<<<65536, 256, 0, stream>>>(x_in, cw1, cb1, raw);
  bn_stats_k<<<32, 256, 0, stream>>>(raw, stats, 32, 64 * 1024);
  bn_gelu_k<<<65536, 256, 0, stream>>>(raw, stats, actA, 31, 16, 0);

  conv_wmma_k<5, 2><<<8192, 256, 0, stream>>>(wc2, actA, cb2, raw, 64, 32, 800, 64, 32, 16384);
  bn_stats_k<<<64, 256, 0, stream>>>(raw, stats, 64, 32 * 1024);
  bn_gelu_k<<<65536, 256, 0, stream>>>(raw, stats, actB, 63, 15, 0);

  conv_wmma_k<7, 3><<<8192, 256, 0, stream>>>(wc3, actB, cb3, raw, 128, 64, 3136, 32, 16, 8192);
  bn_stats_k<<<128, 256, 0, stream>>>(raw, stats, 128, 16 * 1024);
  bn_gelu_k<<<65536, 256, 0, stream>>>(raw, stats, actA, 127, 14, 0);

  conv_wmma_k<7, 3><<<4096, 256, 0, stream>>>(wc4, actA, cb4, raw, 128, 128, 6272, 16, 8, 4096);
  bn_stats_k<<<128, 256, 0, stream>>>(raw, stats, 128, 8 * 1024);
  bn_gelu_k<<<32768, 256, 0, stream>>>(raw, stats, actB, 127, 13, 1);   // -> tokens in actB

  // ---- projection + positional encoding (M=8192, N=256, K=1024) ----
  gemm_wmma_k<<<256, 256, 0, stream>>>(actB, projbt, proj_b, xres, nullptr,
                                       nullptr, nullptr, nullptr, 8192, 256, 1024, 4);

  // ---- transformer blocks ----
  const int mtypes[4] = {0, 1, 2, 2};
  for (int k = 0; k < 4; ++k){
    layernorm_k<<<8192, 256, 0, stream>>>(xres, ln1_g + k * 256, ln1_b + k * 256, xn);
    gemm_wmma_k<<<768, 256, 0, stream>>>(xn, qkvbt + (size_t)k * 768 * 256, bqkv + k * 768,
                                         nullptr, nullptr, qb, kbuf, vbuf,
                                         8192, 768, 256, 3);
    for (int b = 0; b < 8; ++b){
      const u16t* qB = qb   + (size_t)b * 8 * 1024 * 32;
      const u16t* kB = kbuf + (size_t)b * 8 * 1024 * 32;
      const u16t* vB = vbuf + (size_t)b * 8 * 1024 * 32;
      attn_scores_k<<<4096, 256, 0, stream>>>(qB, kB, scores, mtypes[k]);
      softmax_k<<<8192, 256, 0, stream>>>(scores, probs);
      attn_av_k<<<8, 256, 0, stream>>>(probs, vB, obf, b * 1024);
    }
    gemm_wmma_k<<<256, 256, 0, stream>>>(obf, wobt + (size_t)k * 256 * 256, bo + k * 256,
                                         xres, nullptr, nullptr, nullptr, nullptr,
                                         8192, 256, 256, 1);
    layernorm_k<<<8192, 256, 0, stream>>>(xres, ln2_g + k * 256, ln2_b + k * 256, xn);
    gemm_wmma_k<<<1024, 256, 0, stream>>>(xn, w1bt + (size_t)k * 1024 * 256, b1 + k * 1024,
                                          nullptr, hbuf, nullptr, nullptr, nullptr,
                                          8192, 1024, 256, 2);
    gemm_wmma_k<<<256, 256, 0, stream>>>(hbuf, w2bt + (size_t)k * 256 * 1024, b2 + k * 256,
                                         xres, nullptr, nullptr, nullptr, nullptr,
                                         8192, 256, 1024, 1);
  }

  hipMemcpyAsync(d_out, xres, (size_t)out_size * sizeof(float),
                 hipMemcpyDeviceToDevice, stream);
}
